// SelfAttention_67448166416935
// MI455X (gfx1250) — matrix operand order repaired and checked by reading, not compiled, not run on hardware
//
#include <hip/hip_runtime.h>
#include <hip/hip_bf16.h>
#include <stdint.h>

#define B_    4
#define S_    2048
#define E_    1024
#define H_    16
#define D_    64
#define NTOK  (B_ * S_)          // 8192 tokens

typedef __bf16 bf16_t;
typedef __bf16 v16bf __attribute__((ext_vector_type(16)));
typedef float  v8f   __attribute__((ext_vector_type(8)));

union Frag16 { v16bf v; uint4 q[2]; };

// ---------- helpers ----------

__device__ __forceinline__ bf16_t f32_to_bf16(float f) {
  uint32_t u = __builtin_bit_cast(uint32_t, f);
  uint32_t r = u + 0x7FFFu + ((u >> 16) & 1u);   // round-to-nearest-even
  unsigned short hs = (unsigned short)(r >> 16);
  return __builtin_bit_cast(bf16_t, hs);
}

// B-operand fragment: lane holds one column, 16 contiguous K values
__device__ __forceinline__ v16bf load16(const bf16_t* p) {
  Frag16 f;
  f.q[0] = *(const uint4*)(p);
  f.q[1] = *(const uint4*)(p + 8);
  return f.v;
}

// A-operand fragment: lane holds one row; K layout = [k0..k0+15]
__device__ __forceinline__ v16bf loadA(const bf16_t* row, int k0) {
  Frag16 f;
  f.q[0] = *(const uint4*)(row + k0);
  f.q[1] = *(const uint4*)(row + k0 + 8);
  return f.v;
}

__device__ __forceinline__ v8f wmma_bf16(v16bf a, v16bf b, v8f c) {
  return __builtin_amdgcn_wmma_f32_16x16x32_bf16(false, a, false, b, (short)0, c,
                                                 false, false);
}

// CDNA5 async copy: global -> LDS, 16 bytes, tracked by ASYNCcnt
__device__ __forceinline__ void async_copy16(void* lds_dst, const void* src) {
  uint32_t l = (uint32_t)(uintptr_t)lds_dst;   // low 32 bits = LDS offset
  asm volatile("global_load_async_to_lds_b128 %0, %1, off"
               :: "v"(l), "v"(src) : "memory");
}
__device__ __forceinline__ void wait_async() {
  asm volatile("s_wait_asynccnt 0" ::: "memory");
}
__device__ __forceinline__ void wait_ds() {
  asm volatile("s_wait_dscnt 0" ::: "memory");
}

// reductions within a 16-lane half (C-fragment rows live across 16 lanes)
__device__ __forceinline__ float half_max(float v) {
  #pragma unroll
  for (int m = 1; m < 16; m <<= 1) v = fmaxf(v, __shfl_xor(v, m, 32));
  return v;
}
__device__ __forceinline__ float half_sum(float v) {
  #pragma unroll
  for (int m = 1; m < 16; m <<= 1) v += __shfl_xor(v, m, 32);
  return v;
}

// ---------- kernel 1: fp32 -> bf16 ----------

__global__ __launch_bounds__(256) void cvt_f32_bf16(const float* __restrict__ src,
                                                    bf16_t* __restrict__ dst, int n) {
  int i = blockIdx.x * 256 + threadIdx.x;
  if (i < n) dst[i] = f32_to_bf16(src[i]);
}

// ---------- epilogue store helpers ----------

__device__ __forceinline__ void qkv_store_frag(v8f acc, int mbase, int n,
                                               const float* __restrict__ bias,
                                               bf16_t* __restrict__ Qo,
                                               bf16_t* __restrict__ Ko,
                                               bf16_t* __restrict__ Vo,
                                               int h8) {
  float bv = bias[n];
  int part = n >> 10;                 // 0:Q 1:K 2:V (uniform per fragment)
  int e    = n & 1023;
  int h    = e >> 6;
  int d    = e & 63;
  bf16_t* dst = (part == 0) ? Qo : (part == 1) ? Ko : Vo;
  #pragma unroll
  for (int r = 0; r < 8; ++r) {
    int m = mbase + r + h8;           // token index
    int b = m >> 11, s = m & 2047;
    dst[((size_t)((b * H_ + h) * S_ + s)) * D_ + d] = f32_to_bf16(acc[r] + bv);
  }
}

__device__ __forceinline__ void out_store_frag(v8f acc, int mbase, int n,
                                               const float* __restrict__ bias,
                                               float* __restrict__ out, int h8) {
  float bv = bias[n];
  #pragma unroll
  for (int r = 0; r < 8; ++r) {
    int m = mbase + r + h8;
    out[(size_t)m * E_ + n] = acc[r] + bv;
  }
}

// ---------- kernel 2: QKV projection ----------
// C[8192,3072] = X[8192,1024] @ Wqkv[3072,1024]^T + b ; scatter to Q/K/V [B,H,S,D] bf16
// 32x64 C tile per wave: B fragments reused across two A-row fragments.

__global__ __launch_bounds__(256) void qkv_gemm(const bf16_t* __restrict__ X,
                                                const bf16_t* __restrict__ W,
                                                const float*  __restrict__ bias,
                                                bf16_t* __restrict__ Qo,
                                                bf16_t* __restrict__ Ko,
                                                bf16_t* __restrict__ Vo) {
  const int lane = threadIdx.x & 31;
  const int wid  = threadIdx.x >> 5;
  const int tile = blockIdx.x * 8 + wid;    // 12288 tiles of 32x64
  const int mt   = tile & 255;              // 256 row tiles (32 rows)
  const int nt   = tile >> 8;               // 48 col tiles (64 cols)
  const int lo = lane & 15, hi = lane >> 4;
  const int h8 = hi * 8, h16 = hi * 16;

  const bf16_t* xrow0 = X + (size_t)(mt * 32 + lo) * E_;
  const bf16_t* xrow1 = X + (size_t)(mt * 32 + 16 + lo) * E_;

  const bf16_t* w0 = W + (size_t)(nt * 64 +  0 + lo) * E_ + h16;
  const bf16_t* w1 = W + (size_t)(nt * 64 + 16 + lo) * E_ + h16;
  const bf16_t* w2 = W + (size_t)(nt * 64 + 32 + lo) * E_ + h16;
  const bf16_t* w3 = W + (size_t)(nt * 64 + 48 + lo) * E_ + h16;

  v8f c00 = {}, c01 = {}, c02 = {}, c03 = {};
  v8f c10 = {}, c11 = {}, c12 = {}, c13 = {};

  for (int kc = 0; kc < E_ / 32; ++kc) {
    v16bf a0 = loadA(xrow0, kc * 32 + h16);
    v16bf a1 = loadA(xrow1, kc * 32 + h16);
    v16bf b0 = load16(w0 + kc * 32);
    v16bf b1 = load16(w1 + kc * 32);
    v16bf b2 = load16(w2 + kc * 32);
    v16bf b3 = load16(w3 + kc * 32);
    c00 = wmma_bf16(a0, b0, c00);  c10 = wmma_bf16(a1, b0, c10);
    c01 = wmma_bf16(a0, b1, c01);  c11 = wmma_bf16(a1, b1, c11);
    c02 = wmma_bf16(a0, b2, c02);  c12 = wmma_bf16(a1, b2, c12);
    c03 = wmma_bf16(a0, b3, c03);  c13 = wmma_bf16(a1, b3, c13);
  }

  const int m0 = mt * 32, m1 = mt * 32 + 16;
  const int nb0 = nt * 64 + lo;
  qkv_store_frag(c00, m0, nb0 +  0, bias, Qo, Ko, Vo, h8);
  qkv_store_frag(c01, m0, nb0 + 16, bias, Qo, Ko, Vo, h8);
  qkv_store_frag(c02, m0, nb0 + 32, bias, Qo, Ko, Vo, h8);
  qkv_store_frag(c03, m0, nb0 + 48, bias, Qo, Ko, Vo, h8);
  qkv_store_frag(c10, m1, nb0 +  0, bias, Qo, Ko, Vo, h8);
  qkv_store_frag(c11, m1, nb0 + 16, bias, Qo, Ko, Vo, h8);
  qkv_store_frag(c12, m1, nb0 + 32, bias, Qo, Ko, Vo, h8);
  qkv_store_frag(c13, m1, nb0 + 48, bias, Qo, Ko, Vo, h8);
}

// ---------- kernel 3: causal flash attention ----------
// grid = B*H * (S/64); 128 threads = 4 waves, each wave owns 16 q-rows.

__global__ __launch_bounds__(128) void attn_fa(const bf16_t* __restrict__ Q,
                                               const bf16_t* __restrict__ K,
                                               const bf16_t* __restrict__ V,
                                               bf16_t* __restrict__ O,
                                               const int* __restrict__ causal_p) {
  __shared__ bf16_t ldsK[32 * 64];        // K block, [key][d]
  __shared__ bf16_t ldsV[64 * 32];        // V block transposed, [d][key]
  __shared__ bf16_t ldsP[4 * 16 * 32];    // per-wave P tile, [row][key]

  const int causal = *causal_p;
  const int tid  = threadIdx.x;
  const int lane = tid & 31, wid = tid >> 5;
  const int bh = blockIdx.x >> 5, qt = blockIdx.x & 31;
  const int q0 = qt * 64, wq0 = q0 + wid * 16;
  const int lo = lane & 15, hi = lane >> 4;
  const int h8 = hi * 8, h16 = hi * 16;

  const bf16_t* Qb = Q + (size_t)bh * S_ * D_;
  const bf16_t* Kb = K + (size_t)bh * S_ * D_;
  const bf16_t* Vb = V + (size_t)bh * S_ * D_;
  bf16_t* ldsPw = ldsP + wid * 16 * 32;

  // Q A-fragments for this wave's 16 rows (D = 64 => two K-chunks of 32)
  const bf16_t* qrow = Qb + (size_t)(wq0 + lo) * D_;
  v16bf qa0 = loadA(qrow, h16);
  v16bf qa1 = loadA(qrow, 32 + h16);

  v8f o0 = {}, o1 = {}, o2 = {}, o3 = {};
  float m_run[8], l_run[8];
  #pragma unroll
  for (int r = 0; r < 8; ++r) { m_run[r] = -__builtin_inff(); l_run[r] = 0.f; }

  const int nkb = causal ? (q0 + 64) / 32 : S_ / 32;
  const float scale = 0.125f;             // 1/sqrt(64)

  for (int jb = 0; jb < nkb; ++jb) {
    const int kb = jb * 32;
    __syncthreads();                      // previous compute done before restage

    // K block: async global->LDS (2 x 16B per thread)
    for (int t2 = tid; t2 < 256; t2 += 128) {
      int kk = t2 >> 3, dc = (t2 & 7) * 8;
      async_copy16(&ldsK[kk * 64 + dc], Kb + (size_t)(kb + kk) * D_ + dc);
    }
    // V block: load + transpose into LDS
    for (int t2 = tid; t2 < 256; t2 += 128) {
      int kk = t2 >> 3, dc = (t2 & 7) * 8;
      union { uint4 q; bf16_t h[8]; } u;
      u.q = *(const uint4*)(Vb + (size_t)(kb + kk) * D_ + dc);
      #pragma unroll
      for (int i = 0; i < 8; ++i) ldsV[(dc + i) * 32 + kk] = u.h[i];
    }
    wait_async();
    __syncthreads();

    if (!causal || kb <= wq0 + 15) {      // wave-uniform: EXEC stays all-ones
      // scores: Q(16x64) x K^T(64x32) -> two 16x16 fp32 fragments
      v8f s0 = {}, s1 = {};
      s0 = wmma_bf16(qa0, load16(&ldsK[(lo)      * 64 + h16]),      s0);
      s1 = wmma_bf16(qa0, load16(&ldsK[(16 + lo) * 64 + h16]),      s1);
      s0 = wmma_bf16(qa1, load16(&ldsK[(lo)      * 64 + 32 + h16]), s0);
      s1 = wmma_bf16(qa1, load16(&ldsK[(16 + lo) * 64 + 32 + h16]), s1);

      #pragma unroll
      for (int r = 0; r < 8; ++r) {
        int row = wq0 + r + h8;
        float x0 = s0[r] * scale, x1 = s1[r] * scale;
        if (causal) {
          if (kb + lo > row)      x0 = -__builtin_inff();
          if (kb + 16 + lo > row) x1 = -__builtin_inff();
        }
        float mx    = half_max(fmaxf(x0, x1));
        float mnew  = fmaxf(m_run[r], mx);
        float alpha = __expf(m_run[r] - mnew);
        float p0 = __expf(x0 - mnew), p1 = __expf(x1 - mnew);
        m_run[r] = mnew;
        l_run[r] = l_run[r] * alpha + p0 + p1;
        o0[r] *= alpha; o1[r] *= alpha; o2[r] *= alpha; o3[r] *= alpha;
        // stash P in per-wave LDS to transpose C-layout -> A-layout
        ldsPw[(r + h8) * 32 + lo]      = f32_to_bf16(p0);
        ldsPw[(r + h8) * 32 + 16 + lo] = f32_to_bf16(p1);
      }
      wait_ds();

      // out += P(16x32) x V(32x64)
      v16bf pa = loadA(ldsPw + lo * 32, h16);
      o0 = wmma_bf16(pa, load16(&ldsV[(lo)      * 32 + h16]), o0);
      o1 = wmma_bf16(pa, load16(&ldsV[(16 + lo) * 32 + h16]), o1);
      o2 = wmma_bf16(pa, load16(&ldsV[(32 + lo) * 32 + h16]), o2);
      o3 = wmma_bf16(pa, load16(&ldsV[(48 + lo) * 32 + h16]), o3);
    }
  }

  const int bz = bh >> 4, hh = bh & 15;
  #pragma unroll
  for (int r = 0; r < 8; ++r) {
    float inv = 1.0f / half_sum(l_run[r]);
    size_t t = (size_t)bz * S_ + (wq0 + r + h8);
    bf16_t* dst = O + t * E_ + hh * D_;
    dst[ 0 + lo] = f32_to_bf16(o0[r] * inv);
    dst[16 + lo] = f32_to_bf16(o1[r] * inv);
    dst[32 + lo] = f32_to_bf16(o2[r] * inv);
    dst[48 + lo] = f32_to_bf16(o3[r] * inv);
  }
}

// ---------- kernel 4: output projection ----------
// out[8192,1024] = A[8192,1024] @ Wout[1024,1024]^T + b  (fp32 store)
// 32x64 C tile per wave.

__global__ __launch_bounds__(256) void out_gemm(const bf16_t* __restrict__ A,
                                                const bf16_t* __restrict__ W,
                                                const float*  __restrict__ bias,
                                                float* __restrict__ out) {
  const int lane = threadIdx.x & 31;
  const int wid  = threadIdx.x >> 5;
  const int tile = blockIdx.x * 8 + wid;    // 4096 tiles of 32x64
  const int mt   = tile & 255;              // 256 row tiles
  const int nt   = tile >> 8;               // 16 col tiles
  const int lo = lane & 15, hi = lane >> 4;
  const int h8 = hi * 8, h16 = hi * 16;

  const bf16_t* arow0 = A + (size_t)(mt * 32 + lo) * E_;
  const bf16_t* arow1 = A + (size_t)(mt * 32 + 16 + lo) * E_;

  const bf16_t* w0 = W + (size_t)(nt * 64 +  0 + lo) * E_ + h16;
  const bf16_t* w1 = W + (size_t)(nt * 64 + 16 + lo) * E_ + h16;
  const bf16_t* w2 = W + (size_t)(nt * 64 + 32 + lo) * E_ + h16;
  const bf16_t* w3 = W + (size_t)(nt * 64 + 48 + lo) * E_ + h16;

  v8f c00 = {}, c01 = {}, c02 = {}, c03 = {};
  v8f c10 = {}, c11 = {}, c12 = {}, c13 = {};

  for (int kc = 0; kc < E_ / 32; ++kc) {
    v16bf a0 = loadA(arow0, kc * 32 + h16);
    v16bf a1 = loadA(arow1, kc * 32 + h16);
    v16bf b0 = load16(w0 + kc * 32);
    v16bf b1 = load16(w1 + kc * 32);
    v16bf b2 = load16(w2 + kc * 32);
    v16bf b3 = load16(w3 + kc * 32);
    c00 = wmma_bf16(a0, b0, c00);  c10 = wmma_bf16(a1, b0, c10);
    c01 = wmma_bf16(a0, b1, c01);  c11 = wmma_bf16(a1, b1, c11);
    c02 = wmma_bf16(a0, b2, c02);  c12 = wmma_bf16(a1, b2, c12);
    c03 = wmma_bf16(a0, b3, c03);  c13 = wmma_bf16(a1, b3, c13);
  }

  const int m0 = mt * 32, m1 = mt * 32 + 16;
  const int nb0 = nt * 64 + lo;
  out_store_frag(c00, m0, nb0 +  0, bias, out, h8);
  out_store_frag(c01, m0, nb0 + 16, bias, out, h8);
  out_store_frag(c02, m0, nb0 + 32, bias, out, h8);
  out_store_frag(c03, m0, nb0 + 48, bias, out, h8);
  out_store_frag(c10, m1, nb0 +  0, bias, out, h8);
  out_store_frag(c11, m1, nb0 + 16, bias, out, h8);
  out_store_frag(c12, m1, nb0 + 32, bias, out, h8);
  out_store_frag(c13, m1, nb0 + 48, bias, out, h8);
}

// ---------- host launcher ----------

extern "C" void kernel_launch(void* const* d_in, const int* in_sizes, int n_in,
                              void* d_out, int out_size, void* d_ws, size_t ws_size,
                              hipStream_t stream) {
  (void)in_sizes; (void)n_in; (void)out_size; (void)ws_size;
  const float* x    = (const float*)d_in[0];
  const float* wqkv = (const float*)d_in[1];
  const float* bqkv = (const float*)d_in[2];
  const float* wout = (const float*)d_in[3];
  const float* bout = (const float*)d_in[4];
  const int*   causal = (const int*)d_in[5];
  float* out = (float*)d_out;

  char* ws = (char*)d_ws;
  size_t off = 0;
  auto alloc = [&](size_t bytes) {
    void* p = ws + off;
    off += (bytes + 255) & ~(size_t)255;
    return p;
  };
  bf16_t* Xb = (bf16_t*)alloc((size_t)NTOK * E_ * 2);       // 16 MB
  bf16_t* Wq = (bf16_t*)alloc((size_t)3 * E_ * E_ * 2);     //  6 MB
  bf16_t* Wo = (bf16_t*)alloc((size_t)E_ * E_ * 2);         //  2 MB
  bf16_t* Qb = (bf16_t*)alloc((size_t)NTOK * E_ * 2);       // 16 MB
  bf16_t* Kb = (bf16_t*)alloc((size_t)NTOK * E_ * 2);       // 16 MB
  bf16_t* Vb = (bf16_t*)alloc((size_t)NTOK * E_ * 2);       // 16 MB
  bf16_t* Ab = (bf16_t*)alloc((size_t)NTOK * E_ * 2);       // 16 MB

  int n1 = NTOK * E_;
  cvt_f32_bf16<<<(n1 + 255) / 256, 256, 0, stream>>>(x, Xb, n1);
  int n2 = 3 * E_ * E_;
  cvt_f32_bf16<<<(n2 + 255) / 256, 256, 0, stream>>>(wqkv, Wq, n2);
  int n3 = E_ * E_;
  cvt_f32_bf16<<<(n3 + 255) / 256, 256, 0, stream>>>(wout, Wo, n3);

  qkv_gemm<<<1536, 256, 0, stream>>>(Xb, Wq, bqkv, Qb, Kb, Vb);
  attn_fa<<<2048, 128, 0, stream>>>(Qb, Kb, Vb, Ab, causal);
  out_gemm<<<512, 256, 0, stream>>>(Ab, Wo, bout, out);
}
